// Bond2AtomBlock_3736621548058
// MI455X (gfx1250) — compile-verified, matrix operand order checked by reading, not run on hardware
//
#include <hip/hip_runtime.h>
#include <cstddef>

#define HDIM 128
#define D1   129
#define KX   384            // 3*H concat width
#define NT_E 9              // output col tiles (144 cols, 129 used)
#define KC_E 12             // K chunks of 32
#define XSTR 392            // x-tile row stride (halves), padded vs 384 for banks
#define OSTR 148            // out-tile row stride (floats); slot 144 = gate
#define NT_A 8
#define KC_A 4
#define ASTR 136            // atom staging row stride (halves)
#define WAVES 8
#define TPB   256

typedef __attribute__((ext_vector_type(16))) _Float16     v16h;
typedef __attribute__((ext_vector_type(8)))  float        v8f;
typedef __attribute__((ext_vector_type(4)))  unsigned int u32x4;

union HFrag { v16h h; u32x4 q[2]; };

__device__ __forceinline__ void store_h8(_Float16* p, float4 a, float4 b) {
  union { _Float16 h[8]; u32x4 q; } c;
  c.h[0] = (_Float16)a.x; c.h[1] = (_Float16)a.y;
  c.h[2] = (_Float16)a.z; c.h[3] = (_Float16)a.w;
  c.h[4] = (_Float16)b.x; c.h[5] = (_Float16)b.y;
  c.h[6] = (_Float16)b.z; c.h[7] = (_Float16)b.w;
  *(u32x4*)p = c.q;                      // one ds_store_b128
}

// Double-buffered 16xK @ KxN WMMA tile GEMM.
// A staged row-major f16 in LDS (stride XS halves), B pre-packed per-lane
// contiguous fragments. Global wmma-index parity rotates the two B buffers
// correctly across kc boundaries (works for odd and even NT).
template <int NT, int KC, int XS>
__device__ __forceinline__ void gemm_tile(v8f (&acc)[NT], const _Float16* aBase,
                                          const _Float16* wBase, int lane,
                                          int l16, int lhalf) {
  HFrag a, an, b0, b1;
  auto lda = [&](HFrag& f, int kc) {
    const _Float16* ap = aBase + l16 * XS + kc * 32 + lhalf * 8;
    f.q[0] = *(const u32x4*)ap;
    f.q[1] = *(const u32x4*)(ap + 16);
  };
  auto ldb = [&](HFrag& f, int nt, int kc) {
    const _Float16* bp = wBase + ((nt * KC + kc) * 32 + lane) * 16;
    f.q[0] = *(const u32x4*)bp;
    f.q[1] = *(const u32x4*)(bp + 8);
  };
  lda(a, 0);
  ldb(b0, 0, 0);
#pragma unroll
  for (int kc = 0; kc < KC; ++kc) {
    if (kc + 1 < KC) lda(an, kc + 1);          // prefetch next A chunk
#pragma unroll
    for (int nt = 0; nt < NT; ++nt) {
      const int m = kc * NT + nt;
      HFrag& bc = (m & 1) ? b1 : b0;
      HFrag& bn = (m & 1) ? b0 : b1;
      if (nt + 1 < NT)      ldb(bn, nt + 1, kc);     // prefetch next B
      else if (kc + 1 < KC) ldb(bn, 0, kc + 1);
      acc[nt] = __builtin_amdgcn_wmma_f32_16x16x32_f16(
          false, a.h, false, bc.h, (short)0, acc[nt], false, false);
    }
    if (kc + 1 < KC) a = an;
  }
}

// map (k,n) of a row-major [K x N] weight into the WMMA f16 B-fragment pack:
// element e of lane L of tile (kc,nt) stored contiguously (16 halves / lane).
__device__ __forceinline__ int pack_idx(int k, int n, int kcCount) {
  int kc  = k >> 5, kr = k & 31;
  int e   = (kr & 7) | (((kr >> 4) & 1) << 3);
  int Lhi = (kr >> 3) & 1;
  int L   = (Lhi << 4) | (n & 15);
  int nt  = n >> 4;
  return ((nt * kcCount + kc) * 32 + L) * 16 + e;
}

// ---------- prep 1: fold BN scales into W1, W2 (f32 scratch) ----------
__global__ void k_prep1(const float* __restrict__ W1, const float* __restrict__ g1,
                        const float* __restrict__ v1,
                        const float* __restrict__ W2, const float* __restrict__ g2,
                        const float* __restrict__ v2,
                        float* __restrict__ W1s, float* __restrict__ W2s) {
  int i = blockIdx.x * blockDim.x + threadIdx.x;
  if (i < KX * D1) {
    int j = i % D1;
    W1s[i] = W1[i] * (g1[j] * rsqrtf(v1[j] + 1e-3f));
  }
  int k = i - KX * D1;
  if (k >= 0 && k < D1 * D1) {
    int j = k % D1;
    W2s[k] = W2[k] * (g2[j] * rsqrtf(v2[j] + 1e-3f));
  }
}

// ---------- prep 2: Weff = W1s @ W2s (packed f16) and beff ----------
__global__ void k_prep2(const float* __restrict__ W1s, const float* __restrict__ W2s,
                        const float* __restrict__ b1, const float* __restrict__ m1,
                        const float* __restrict__ g1, const float* __restrict__ v1,
                        const float* __restrict__ b2, const float* __restrict__ m2,
                        const float* __restrict__ g2, const float* __restrict__ v2,
                        _Float16* __restrict__ wEff, float* __restrict__ bEff) {
  int i = blockIdx.x * blockDim.x + threadIdx.x;
  if (i >= KX * (NT_E * 16)) return;
  int k = i / (NT_E * 16), n = i % (NT_E * 16);
  float acc = 0.0f;
  if (n < D1)
    for (int j = 0; j < D1; ++j) acc += W1s[k * D1 + j] * W2s[j * D1 + n];
  wEff[pack_idx(k, n, KC_E)] = (_Float16)acc;
  if (i < NT_E * 16) {            // fused bias (one thread per output col)
    float bv = 0.0f;
    if (i < D1) {
      float s2 = g2[i] * rsqrtf(v2[i] + 1e-3f);
      bv = b2[i] - m2[i] * s2;
      for (int j = 0; j < D1; ++j) {
        float s1 = g1[j] * rsqrtf(v1[j] + 1e-3f);
        float b1e = b1[j] - m1[j] * s1;
        bv += b1e * W2s[j * D1 + i];
      }
    }
    bEff[i] = bv;
  }
}

// ---------- prep 3: pack the 4 residual weight matrices (f16 B-fragments) ----
__global__ void k_prep3(const float* __restrict__ w0, const float* __restrict__ w1,
                        const float* __restrict__ w2, const float* __restrict__ w3,
                        _Float16* __restrict__ dst) {
  int i = blockIdx.x * blockDim.x + threadIdx.x;
  if (i >= 4 * HDIM * HDIM) return;
  int mat = i / (HDIM * HDIM), r = i % (HDIM * HDIM);
  int k = r / HDIM, n = r % HDIM;
  const float* W = (mat == 0) ? w0 : (mat == 1) ? w1 : (mat == 2) ? w2 : w3;
  dst[mat * (NT_A * KC_A * 32 * 16) + pack_idx(k, n, KC_A)] = (_Float16)W[r];
}

// ---------- init: d_out = atom_embedding (accumulator base) ----------
__global__ void k_init(const float* __restrict__ src, float* __restrict__ dst, int n) {
  int i = blockIdx.x * blockDim.x + threadIdx.x;
  if (i < n) dst[i] = src[i];
}

// ---------- edge kernel: fused affine + gate + atomic scatter ----------
__global__ __launch_bounds__(TPB) void k_edge(
    const float* __restrict__ atomEmb, const float* __restrict__ bondEmb,
    const int* __restrict__ idxI, const int* __restrict__ idxJ,
    const _Float16* __restrict__ wEff, const float* __restrict__ bEff,
    float* __restrict__ atomAcc, int nTiles) {
  __shared__ _Float16 sW[NT_E * KC_E * 32 * 16];           // 110,592 B
  __shared__ float    sB[NT_E * 16];
  __shared__ __align__(16) unsigned char sScr[WAVES][16 * XSTR * 2];  // 12,544 B/wave

  const int tid = threadIdx.x;
  { // cooperative persistent weight load
    const u32x4* src = (const u32x4*)wEff;
    u32x4* dst = (u32x4*)sW;
    for (int i = tid; i < NT_E * KC_E * 32 * 2; i += TPB) dst[i] = src[i];
    for (int i = tid; i < NT_E * 16; i += TPB) sB[i] = bEff[i];
  }
  __syncthreads();

  const int wave = tid >> 5, lane = tid & 31;
  const int lhalf = lane >> 4, l16 = lane & 15;
  _Float16* xw = (_Float16*)sScr[wave];
  float*    ow = (float*)sScr[wave];

  const int wg = blockIdx.x * WAVES + wave;
  const int wc = gridDim.x * WAVES;

  for (int t = wg; t < nTiles; t += wc) {
    const int e0 = t * 16;

    // prefetch next tile's bond rows (pure HBM stream, 8KB) into caches
    if (t + wc < nTiles) {
      const char* nb = (const char*)(bondEmb + (size_t)(t + wc) * 16 * HDIM);
      __builtin_prefetch(nb + lane * 128, 0, 1);
      __builtin_prefetch(nb + 4096 + lane * 128, 0, 1);
    }

    // ---- stage 16 x 384 f16 x-tile: [ai | bond | aj] ----
#pragma unroll 2
    for (int it = 0; it < 8; ++it) {
      int row = it * 2 + lhalf;
      int gi = idxI[e0 + row];
      int gj = idxJ[e0 + row];
      const float4* pa = (const float4*)(atomEmb + (size_t)gi * HDIM + l16 * 8);
      const float4* pb = (const float4*)(bondEmb + (size_t)(e0 + row) * HDIM + l16 * 8);
      const float4* pc = (const float4*)(atomEmb + (size_t)gj * HDIM + l16 * 8);
      float4 a0 = pa[0], a1 = pa[1];
      float4 b0 = pb[0], b1 = pb[1];
      float4 c0 = pc[0], c1 = pc[1];
      store_h8(xw + row * XSTR +   0 + l16 * 8, a0, a1);
      store_h8(xw + row * XSTR + 128 + l16 * 8, b0, b1);
      store_h8(xw + row * XSTR + 256 + l16 * 8, c0, c1);
    }
    // (same-wave DS ops are in-order: no barrier needed)

    // ---- 16x384 @ 384x144 via v_wmma_f32_16x16x32_f16 ----
    v8f acc[NT_E] = {};
    gemm_tile<NT_E, KC_E, XSTR>(acc, xw, sW, lane, l16, lhalf);

    // ---- bias, spill to LDS f32; col0 (gate) -> slot 144, col n -> n-1 ----
#pragma unroll
    for (int nt = 0; nt < NT_E; ++nt) {
      int col = nt * 16 + l16;
      int slot = col ? (col - 1) : 144;
      float bias = sB[col];
#pragma unroll
      for (int r = 0; r < 8; ++r) {
        int row = r + lhalf * 8;
        ow[row * OSTR + slot] = acc[nt][r] + bias;
      }
    }
    // ---- gate * message, aligned float4 reads + atomic scatter ----
    {
      int row = lane >> 1;
      int cb = (lane & 1) * 64;
      float gate = ow[row * OSTR + 144];
      int ai = idxI[e0 + row];
      float* dst = atomAcc + (size_t)ai * HDIM + cb;
      const float4* src4 = (const float4*)(ow + row * OSTR + cb);
#pragma unroll
      for (int v = 0; v < 16; ++v) {
        float4 mv = src4[v];
        (void)__hip_atomic_fetch_add(dst + v * 4 + 0, gate * mv.x,
                                     __ATOMIC_RELAXED, __HIP_MEMORY_SCOPE_AGENT);
        (void)__hip_atomic_fetch_add(dst + v * 4 + 1, gate * mv.y,
                                     __ATOMIC_RELAXED, __HIP_MEMORY_SCOPE_AGENT);
        (void)__hip_atomic_fetch_add(dst + v * 4 + 2, gate * mv.z,
                                     __ATOMIC_RELAXED, __HIP_MEMORY_SCOPE_AGENT);
        (void)__hip_atomic_fetch_add(dst + v * 4 + 3, gate * mv.w,
                                     __ATOMIC_RELAXED, __HIP_MEMORY_SCOPE_AGENT);
      }
    }
  }
}

// ---------- atom kernel: two residual MLPs, in-place on d_out ----------
__global__ __launch_bounds__(TPB) void k_atom(
    float* __restrict__ atomAcc, const _Float16* __restrict__ resW,
    const float* __restrict__ rb1a, const float* __restrict__ rb1b,
    const float* __restrict__ rb2a, const float* __restrict__ rb2b,
    int nTiles) {
  __shared__ _Float16 sW[4 * NT_A * KC_A * 32 * 16];        // 131,072 B
  __shared__ __align__(16) unsigned char sScr[WAVES][2 * 16 * ASTR * 2];

  const int tid = threadIdx.x;
  {
    const u32x4* src = (const u32x4*)resW;
    u32x4* dst = (u32x4*)sW;
    for (int i = tid; i < 4 * NT_A * KC_A * 32 * 2; i += TPB) dst[i] = src[i];
  }
  __syncthreads();

  const int wave = tid >> 5, lane = tid & 31;
  const int lhalf = lane >> 4, l16 = lane & 15;
  _Float16* aLds = (_Float16*)sScr[wave];
  _Float16* hLds = aLds + 16 * ASTR;
  const int wg = blockIdx.x * WAVES + wave;
  const int wc = gridDim.x * WAVES;

  for (int t = wg; t < nTiles; t += wc) {
    const int m0 = t * 16;
    v8f a[NT_A];
#pragma unroll
    for (int nt = 0; nt < NT_A; ++nt)
#pragma unroll
      for (int r = 0; r < 8; ++r) {
        int row = r + lhalf * 8;
        float v = atomAcc[(size_t)(m0 + row) * HDIM + nt * 16 + l16];
        a[nt][r] = v;
        aLds[row * ASTR + nt * 16 + l16] = (_Float16)v;
      }

    const float* biases[4] = {rb1a, rb1b, rb2a, rb2b};
#pragma unroll 1
    for (int layer = 0; layer < 2; ++layer) {
      const _Float16* w1 = sW + (layer * 2 + 0) * (NT_A * KC_A * 32 * 16);
      const _Float16* w2 = sW + (layer * 2 + 1) * (NT_A * KC_A * 32 * 16);
      const float* b1 = biases[layer * 2 + 0];
      const float* b2 = biases[layer * 2 + 1];

      v8f h[NT_A] = {};
      gemm_tile<NT_A, KC_A, ASTR>(h, aLds, w1, lane, l16, lhalf);
#pragma unroll
      for (int nt = 0; nt < NT_A; ++nt) {
        float bias = b1[nt * 16 + l16];
#pragma unroll
        for (int r = 0; r < 8; ++r) {
          int row = r + lhalf * 8;
          hLds[row * ASTR + nt * 16 + l16] = (_Float16)(h[nt][r] + bias);
        }
      }
      v8f g[NT_A] = {};
      gemm_tile<NT_A, KC_A, ASTR>(g, hLds, w2, lane, l16, lhalf);
#pragma unroll
      for (int nt = 0; nt < NT_A; ++nt) {
        float bias = b2[nt * 16 + l16];
#pragma unroll
        for (int r = 0; r < 8; ++r) {
          int row = r + lhalf * 8;
          float nv = a[nt][r] + g[nt][r] + bias;   // residual
          a[nt][r] = nv;
          if (layer == 0) aLds[row * ASTR + nt * 16 + l16] = (_Float16)nv;
        }
      }
    }
#pragma unroll
    for (int nt = 0; nt < NT_A; ++nt)
#pragma unroll
      for (int r = 0; r < 8; ++r) {
        int row = r + lhalf * 8;
        atomAcc[(size_t)(m0 + row) * HDIM + nt * 16 + l16] = a[nt][r];
      }
  }
}

extern "C" void kernel_launch(void* const* d_in, const int* in_sizes, int n_in,
                              void* d_out, int out_size, void* d_ws, size_t ws_size,
                              hipStream_t stream) {
  const float* atomEmb = (const float*)d_in[0];
  const float* bondEmb = (const float*)d_in[1];
  const int*   idxI    = (const int*)d_in[2];
  const int*   idxJ    = (const int*)d_in[3];
  const float* W1 = (const float*)d_in[4];
  const float* g1 = (const float*)d_in[5];
  const float* b1 = (const float*)d_in[6];
  const float* m1 = (const float*)d_in[7];
  const float* v1 = (const float*)d_in[8];
  const float* W2 = (const float*)d_in[9];
  const float* g2 = (const float*)d_in[10];
  const float* b2 = (const float*)d_in[11];
  const float* m2 = (const float*)d_in[12];
  const float* v2 = (const float*)d_in[13];
  const float* r1w1 = (const float*)d_in[14];
  const float* r1b1 = (const float*)d_in[15];
  const float* r1w2 = (const float*)d_in[16];
  const float* r1b2 = (const float*)d_in[17];
  const float* r2w1 = (const float*)d_in[18];
  const float* r2b1 = (const float*)d_in[19];
  const float* r2w2 = (const float*)d_in[20];
  const float* r2b2 = (const float*)d_in[21];

  const int nAtoms = in_sizes[0] / HDIM;
  const int nEdges = in_sizes[2];
  float* out = (float*)d_out;

  // workspace layout
  char* ws = (char*)d_ws;
  float*    W1s  = (float*)(ws);                          // 384*129 f32
  float*    W2s  = (float*)(ws + 198144);                 // 129*129 f32
  _Float16* wEff = (_Float16*)(ws + 264720);              // 55,296 halves
  float*    bEff = (float*)(ws + 375312);                 // 144 f32
  _Float16* resW = (_Float16*)(ws + 375888);              // 65,536 halves

  int nPrep1 = KX * D1 + D1 * D1;
  k_prep1<<<(nPrep1 + 255) / 256, 256, 0, stream>>>(W1, g1, v1, W2, g2, v2, W1s, W2s);
  int nPrep2 = KX * NT_E * 16;
  k_prep2<<<(nPrep2 + 255) / 256, 256, 0, stream>>>(W1s, W2s, b1, m1, g1, v1,
                                                    b2, m2, g2, v2, wEff, bEff);
  k_prep3<<<(4 * HDIM * HDIM + 255) / 256, 256, 0, stream>>>(r1w1, r1w2, r2w1, r2w2, resW);
  k_init<<<(nAtoms * HDIM + 255) / 256, 256, 0, stream>>>(atomEmb, out, nAtoms * HDIM);

  k_edge<<<512, TPB, 0, stream>>>(atomEmb, bondEmb, idxI, idxJ, wEff, bEff,
                                  out, nEdges / 16);
  k_atom<<<224, TPB, 0, stream>>>(out, resW, r1b1, r1b2, r2b1, r2b2, nAtoms / 16);
}
